// MoA_29429115912986
// MI455X (gfx1250) — matrix operand order repaired and checked by reading, not compiled, not run on hardware
//
#include <hip/hip_runtime.h>

// ---------------------------------------------------------------------------
// MoA forward for MI455X (gfx1250), wave32 + v_wmma_f32_16x16x32_f16.
// B=4, T=1024, C=1024, E=8, H=16, D=64, TOPK=2.
//
// Pipeline: gate -> K/V proj -> per-expert Q proj (predicated gather) ->
//           flash attention (WMMA scores + WMMA P.V) ->
//           per-expert O proj (predicated scatter) -> weighted combine.
// Projection GEMMs: one wave owns a 64x32 output tile (4 A-frags x 2 B-frags,
// 8 WMMA per K-step, ~10.7 FLOP/B per lane), f16 operands / f32 accumulate.
// ---------------------------------------------------------------------------

typedef __attribute__((ext_vector_type(16))) _Float16 v16h;
typedef __attribute__((ext_vector_type(8)))  float    v8f;

#define BB   4
#define TT   1024
#define CC   1024
#define EE   8
#define HH   16
#define DD   64
#define NROW 4096            // B*T
#define NEG_BIG (-1.0e30f)

static __device__ __forceinline__ v8f wmma_f16(v16h a, v16h b, v8f c) {
  return __builtin_amdgcn_wmma_f32_16x16x32_f16(false, a, false, b, (short)0, c,
                                                false, false);
}

static __device__ __forceinline__ v8f zero8() {
  v8f c;
#pragma unroll
  for (int i = 0; i < 8; ++i) c[i] = 0.0f;
  return c;
}

// A fragment 16x32: lane&15 = M row; slots 0-7 = K half*8+{0..7}, slots 8-15 = +16.
static __device__ __forceinline__ v16h load_a_frag(const float* __restrict__ A,
                                                   int lda, int kk) {
  const int lane = threadIdx.x & 31;
  const float* p = A + (size_t)(lane & 15) * lda + kk + ((lane >> 4) << 3);
  v16h a;
#pragma unroll
  for (int i = 0; i < 8; ++i) a[i] = (_Float16)p[i];
#pragma unroll
  for (int i = 0; i < 8; ++i) a[8 + i] = (_Float16)p[16 + i];
  return a;
}

static __device__ __forceinline__ v16h load_a_frag(const _Float16* __restrict__ A,
                                                   int lda, int kk) {
  const int lane = threadIdx.x & 31;
  const _Float16* p = A + (size_t)(lane & 15) * lda + kk + ((lane >> 4) << 3);
  v16h a;
#pragma unroll
  for (int i = 0; i < 8; ++i) a[i] = p[i];
#pragma unroll
  for (int i = 0; i < 8; ++i) a[8 + i] = p[16 + i];
  return a;
}

// B fragment 32x16: lane&15 = N col; slots 0-7 = K half*8+{0..7}, slots 8-15 = +16.
// B must be pre-offset to column n0 of a row-major KxN matrix.
static __device__ __forceinline__ v16h load_b_f32(const float* __restrict__ B,
                                                  int ldb, int kk) {
  const int lane = threadIdx.x & 31;
  const float* p = B + (size_t)(kk + ((lane >> 4) << 3)) * ldb + (lane & 15);
  v16h b;
#pragma unroll
  for (int i = 0; i < 8; ++i) b[i] = (_Float16)p[(size_t)i * ldb];
#pragma unroll
  for (int i = 0; i < 8; ++i) b[8 + i] = (_Float16)p[(size_t)(16 + i) * ldb];
  return b;
}

static __device__ __forceinline__ v16h load_b_f16(const _Float16* __restrict__ B,
                                                  int ldb, int kk) {
  const int lane = threadIdx.x & 31;
  const _Float16* p = B + (size_t)(kk + ((lane >> 4) << 3)) * ldb + (lane & 15);
  v16h b;
#pragma unroll
  for (int i = 0; i < 8; ++i) b[i] = p[(size_t)i * ldb];
#pragma unroll
  for (int i = 0; i < 8; ++i) b[8 + i] = p[(size_t)(16 + i) * ldb];
  return b;
}

// ---------------------------------------------------------------------------
// Register-blocked 64(M) x 32(N) GEMM body: C += A(64xCC) * B(CCx32).
// A row-major (f32 or f16), B row-major f32 (weights), f32 accumulate.
// ---------------------------------------------------------------------------
template <typename TA>
static __device__ __forceinline__ void gemm_64x32(const TA* __restrict__ A, int lda,
                                                  const float* __restrict__ B, int ldb,
                                                  v8f (&acc)[4][2]) {
  const int lane = threadIdx.x & 31;
#pragma unroll 1
  for (int kk = 0; kk < CC; kk += 32) {
    const v16h b0 = load_b_f32(B, ldb, kk);
    const v16h b1 = load_b_f32(B + 16, ldb, kk);
    if (kk + 32 < CC)  // prefetch next K-slab of the streamed weight operand
      __builtin_prefetch(B + (size_t)(kk + 32 + lane) * ldb, 0, 0);
#pragma unroll
    for (int mi = 0; mi < 4; ++mi) {
      const v16h a = load_a_frag(A + (size_t)(mi * 16) * lda, lda, kk);
      acc[mi][0] = wmma_f16(a, b0, acc[mi][0]);
      acc[mi][1] = wmma_f16(a, b1, acc[mi][1]);
    }
  }
}

// ---------------------------------------------------------------------------
// 1) Gating: logits = x@Wg + noise * softplus(x@Wn); top-2 + softmax.
// One wave per token row.
// ---------------------------------------------------------------------------
__global__ void moa_gate_kernel(const float* __restrict__ x,
                                const float* __restrict__ noise,
                                const float* __restrict__ Wg,
                                const float* __restrict__ Wn,
                                int* __restrict__ sel, float* __restrict__ wgt) {
  const int row  = blockIdx.x;
  const int lane = threadIdx.x;
  const float* xr = x + (size_t)row * CC;
  float ag[EE], an[EE];
#pragma unroll
  for (int e = 0; e < EE; ++e) { ag[e] = 0.0f; an[e] = 0.0f; }
  for (int c = lane; c < CC; c += 32) {
    const float xv = xr[c];
    const float* g = Wg + (size_t)c * EE;
    const float* n = Wn + (size_t)c * EE;
#pragma unroll
    for (int e = 0; e < EE; ++e) { ag[e] += xv * g[e]; an[e] += xv * n[e]; }
  }
#pragma unroll
  for (int e = 0; e < EE; ++e) {
#pragma unroll
    for (int m = 16; m >= 1; m >>= 1) {
      ag[e] += __shfl_xor(ag[e], m, 32);
      an[e] += __shfl_xor(an[e], m, 32);
    }
  }
  if (lane == 0) {
    float logit[EE];
#pragma unroll
    for (int e = 0; e < EE; ++e) {
      const float z  = an[e];
      const float sp = (z > 20.0f) ? z : log1pf(expf(z));
      logit[e] = ag[e] + noise[(size_t)row * EE + e] * sp;
    }
    int i0 = 0; float b0 = logit[0];
#pragma unroll
    for (int e = 1; e < EE; ++e)
      if (logit[e] > b0) { b0 = logit[e]; i0 = e; }
    int i1 = -1; float b1 = NEG_BIG;
#pragma unroll
    for (int e = 0; e < EE; ++e)
      if (e != i0 && logit[e] > b1) { b1 = logit[e]; i1 = e; }
    const float e1 = expf(b1 - b0);
    const float s  = 1.0f + e1;
    sel[row * 2 + 0] = i0;
    sel[row * 2 + 1] = i1;
    wgt[row * 2 + 0] = 1.0f / s;
    wgt[row * 2 + 1] = e1 / s;
  }
}

// ---------------------------------------------------------------------------
// 2) K = x@Wk stored transposed as [b][h][d][t] (f16);  V = x@Wv as [row][c].
// One wave owns a 64x32 output tile.
// ---------------------------------------------------------------------------
__global__ void moa_kv_kernel(const float* __restrict__ x,
                              const float* __restrict__ Wk,
                              const float* __restrict__ Wv,
                              _Float16* __restrict__ KmT,
                              _Float16* __restrict__ Vm) {
  const int m0  = blockIdx.x * 64;
  const int n0  = blockIdx.y * 32;
  const int isv = blockIdx.z;
  v8f acc[4][2];
#pragma unroll
  for (int mi = 0; mi < 4; ++mi) { acc[mi][0] = zero8(); acc[mi][1] = zero8(); }
  gemm_64x32(x + (size_t)m0 * CC, CC, (isv ? Wv : Wk) + n0, CC, acc);

  const int lane = threadIdx.x;
  const int half = lane >> 4;
  if (isv) {
#pragma unroll
    for (int mi = 0; mi < 4; ++mi)
#pragma unroll
      for (int ni = 0; ni < 2; ++ni) {
        const int col = n0 + ni * 16 + (lane & 15);
#pragma unroll
        for (int r = 0; r < 8; ++r) {
          const int row = m0 + mi * 16 + r + (half << 3);
          Vm[(size_t)row * CC + col] = (_Float16)acc[mi][ni][r];
        }
      }
  } else {
    const int b = m0 >> 10;                       // batch of this row tile
#pragma unroll
    for (int mi = 0; mi < 4; ++mi)
#pragma unroll
      for (int ni = 0; ni < 2; ++ni) {
        const int col = n0 + ni * 16 + (lane & 15);
#pragma unroll
        for (int r = 0; r < 8; ++r) {
          const int t = (m0 & (TT - 1)) + mi * 16 + r + (half << 3);
          KmT[((size_t)(b * CC + col)) * TT + t] = (_Float16)acc[mi][ni][r];
        }
      }
  }
}

// ---------------------------------------------------------------------------
// 3) Q: dense per-expert q_tile = x@Wq[e]; predicated f16 store into
//    Qsel[slot] where sel[row][slot]==e. grid.z = expert.
// ---------------------------------------------------------------------------
__global__ void moa_q_kernel(const float* __restrict__ x,
                             const float* __restrict__ Wq,
                             const int* __restrict__ sel,
                             _Float16* __restrict__ Qsel) {
  const int m0 = blockIdx.x * 64;
  const int n0 = blockIdx.y * 32;
  const int e  = blockIdx.z;
  v8f acc[4][2];
#pragma unroll
  for (int mi = 0; mi < 4; ++mi) { acc[mi][0] = zero8(); acc[mi][1] = zero8(); }
  gemm_64x32(x + (size_t)m0 * CC, CC, Wq + (size_t)e * CC * CC + n0, CC, acc);

  const int lane = threadIdx.x;
  const int half = lane >> 4;
#pragma unroll
  for (int mi = 0; mi < 4; ++mi)
#pragma unroll
    for (int ni = 0; ni < 2; ++ni) {
      const int col = n0 + ni * 16 + (lane & 15);
#pragma unroll
      for (int r = 0; r < 8; ++r) {
        const int row = m0 + mi * 16 + r + (half << 3);
        const _Float16 hv = (_Float16)acc[mi][ni][r];
#pragma unroll
        for (int s = 0; s < 2; ++s)
          if (sel[row * 2 + s] == e)
            Qsel[(size_t)s * NROW * CC + (size_t)row * CC + col] = hv;
      }
    }
}

// ---------------------------------------------------------------------------
// 4) Flash attention: one wave = 16 query rows of one (b, h, slot).
//    Scores via WMMA (Q frag x KmT), online softmax, P routed through LDS
//    into an A fragment, P@V via WMMA.
// ---------------------------------------------------------------------------
__global__ void moa_attn_kernel(const _Float16* __restrict__ Qsel,
                                const _Float16* __restrict__ KmT,
                                const _Float16* __restrict__ Vm,
                                _Float16* __restrict__ attn) {
  const int t0   = blockIdx.x * 16;          // query tile within T
  const int bh   = blockIdx.y;               // b*16 + h
  const int slot = blockIdx.z;
  const int b    = bh >> 4;
  const int h    = bh & 15;
  const int lane = threadIdx.x;
  const int half = lane >> 4;

  const _Float16* Q  = Qsel + (size_t)slot * NROW * CC +
                       ((size_t)(b * TT + t0)) * CC + h * DD;
  const _Float16* Kb = KmT + ((size_t)(b * CC + h * DD)) * TT;   // rows = d
  const _Float16* Vb = Vm + ((size_t)(b * TT)) * CC + h * DD;    // rows = j

  const v16h aq0 = load_a_frag(Q, CC, 0);
  const v16h aq1 = load_a_frag(Q, CC, 32);

  v8f o0 = zero8(), o1 = zero8(), o2 = zero8(), o3 = zero8();
  float mrow[8], lrow[8];
#pragma unroll
  for (int r = 0; r < 8; ++r) { mrow[r] = NEG_BIG; lrow[r] = 0.0f; }

  __shared__ _Float16 s_p[16][32];
  const float scale = 0.125f;                // 1/sqrt(64)

  for (int jb = 0; jb <= t0 + 15; jb += 32) {
    // ---- scores S(16x32) = Q(16x64) @ K^T(64x32), two 16-col chunks ----
    v8f s0 = zero8(), s1 = zero8();
    s0 = wmma_f16(aq0, load_b_f16(Kb + jb, TT, 0),       s0);
    s0 = wmma_f16(aq1, load_b_f16(Kb + jb, TT, 32),      s0);
    s1 = wmma_f16(aq0, load_b_f16(Kb + jb + 16, TT, 0),  s1);
    s1 = wmma_f16(aq1, load_b_f16(Kb + jb + 16, TT, 32), s1);

    const int j0 = jb + (lane & 15);
    const int j1 = j0 + 16;
    float alpha[8];
#pragma unroll
    for (int r = 0; r < 8; ++r) {
      const int t = t0 + r + (half << 3);
      float v0 = (j0 <= t) ? s0[r] * scale : NEG_BIG;
      float v1 = (j1 <= t) ? s1[r] * scale : NEG_BIG;
      float mx = fmaxf(v0, v1);
#pragma unroll
      for (int m = 8; m >= 1; m >>= 1) mx = fmaxf(mx, __shfl_xor(mx, m, 32));
      const float mn = fmaxf(mrow[r], mx);
      alpha[r] = expf(mrow[r] - mn);
      const float p0 = expf(v0 - mn);
      const float p1 = expf(v1 - mn);
      s0[r] = p0; s1[r] = p1;
      float rs = p0 + p1;
#pragma unroll
      for (int m = 8; m >= 1; m >>= 1) rs += __shfl_xor(rs, m, 32);
      lrow[r] = lrow[r] * alpha[r] + rs;
      mrow[r] = mn;
    }
#pragma unroll
    for (int r = 0; r < 8; ++r) {
      o0[r] *= alpha[r]; o1[r] *= alpha[r];
      o2[r] *= alpha[r]; o3[r] *= alpha[r];
    }

    // ---- C-layout P -> LDS -> A-layout fragment ----
    __syncthreads();
#pragma unroll
    for (int r = 0; r < 8; ++r) {
      s_p[r + (half << 3)][lane & 15]        = (_Float16)s0[r];
      s_p[r + (half << 3)][16 + (lane & 15)] = (_Float16)s1[r];
    }
    __syncthreads();
    v16h ap;
    {
      const int prow = lane & 15;
      const int kb   = half << 3;
#pragma unroll
      for (int i = 0; i < 8; ++i) ap[i] = s_p[prow][kb + i];
#pragma unroll
      for (int i = 0; i < 8; ++i) ap[8 + i] = s_p[prow][16 + kb + i];
    }

    // ---- O += P(16x32) @ V(32x64) ----
    const _Float16* vrow = Vb + (size_t)jb * CC;
    o0 = wmma_f16(ap, load_b_f16(vrow + 0,  CC, 0), o0);
    o1 = wmma_f16(ap, load_b_f16(vrow + 16, CC, 0), o1);
    o2 = wmma_f16(ap, load_b_f16(vrow + 32, CC, 0), o2);
    o3 = wmma_f16(ap, load_b_f16(vrow + 48, CC, 0), o3);
  }

  _Float16* Op = attn + (size_t)slot * NROW * CC +
                 ((size_t)(b * TT + t0)) * CC + h * DD;
#pragma unroll
  for (int r = 0; r < 8; ++r) {
    const int row = r + (half << 3);
    const float inv = 1.0f / lrow[r];
    Op[(size_t)row * CC + 0  + (lane & 15)] = (_Float16)(o0[r] * inv);
    Op[(size_t)row * CC + 16 + (lane & 15)] = (_Float16)(o1[r] * inv);
    Op[(size_t)row * CC + 32 + (lane & 15)] = (_Float16)(o2[r] * inv);
    Op[(size_t)row * CC + 48 + (lane & 15)] = (_Float16)(o3[r] * inv);
  }
}

// ---------------------------------------------------------------------------
// 5) Output projection: dense per-expert o_tile = attn[slot]@Wo[e];
//    predicated f32 store into osel[slot]. grid.z = e*2 + slot.
// ---------------------------------------------------------------------------
__global__ void moa_oproj_kernel(const _Float16* __restrict__ attn,
                                 const float* __restrict__ Wo,
                                 const int* __restrict__ sel,
                                 float* __restrict__ osel) {
  const int m0   = blockIdx.x * 64;
  const int n0   = blockIdx.y * 32;
  const int e    = blockIdx.z >> 1;
  const int slot = blockIdx.z & 1;
  v8f acc[4][2];
#pragma unroll
  for (int mi = 0; mi < 4; ++mi) { acc[mi][0] = zero8(); acc[mi][1] = zero8(); }
  gemm_64x32(attn + (size_t)slot * NROW * CC + (size_t)m0 * CC, CC,
             Wo + (size_t)e * CC * CC + n0, CC, acc);

  const int lane = threadIdx.x;
  const int half = lane >> 4;
#pragma unroll
  for (int mi = 0; mi < 4; ++mi)
#pragma unroll
    for (int ni = 0; ni < 2; ++ni) {
      const int col = n0 + ni * 16 + (lane & 15);
#pragma unroll
      for (int r = 0; r < 8; ++r) {
        const int row = m0 + mi * 16 + r + (half << 3);
        if (sel[row * 2 + slot] == e)
          osel[(size_t)slot * NROW * CC + (size_t)row * CC + col] = acc[mi][ni][r];
      }
    }
}

// ---------------------------------------------------------------------------
// 6) out = w0*osel0 + w1*osel1
// ---------------------------------------------------------------------------
__global__ void moa_combine_kernel(const float* __restrict__ osel,
                                   const float* __restrict__ wgt,
                                   float* __restrict__ out) {
  const int idx = blockIdx.x * 256 + threadIdx.x;       // exactly NROW*CC
  const int row = idx >> 10;
  out[idx] = wgt[row * 2 + 0] * osel[idx] +
             wgt[row * 2 + 1] * osel[(size_t)NROW * CC + idx];
}

// ---------------------------------------------------------------------------
// Workspace layout (bytes)
// ---------------------------------------------------------------------------
static const size_t OFF_SEL  = 0;                                   // 32 KB
static const size_t OFF_WGT  = 32768;                               // 32 KB
static const size_t OFF_KMT  = 65536;                               // 8 MB f16
static const size_t OFF_VM   = OFF_KMT + (size_t)NROW * CC * 2;     // 8 MB f16
static const size_t OFF_QSEL = OFF_VM + (size_t)NROW * CC * 2;      // 16 MB f16
static const size_t OFF_ATTN = OFF_QSEL + (size_t)2 * NROW * CC * 2;// 16 MB f16
static const size_t OFF_OSEL = OFF_ATTN + (size_t)2 * NROW * CC * 2;// 32 MB f32

extern "C" void kernel_launch(void* const* d_in, const int* in_sizes, int n_in,
                              void* d_out, int out_size, void* d_ws,
                              size_t ws_size, hipStream_t stream) {
  (void)in_sizes; (void)n_in; (void)out_size; (void)ws_size;
  const float* x     = (const float*)d_in[0];
  const float* noise = (const float*)d_in[1];
  const float* Wk    = (const float*)d_in[2];
  const float* Wv    = (const float*)d_in[3];
  const float* Wq    = (const float*)d_in[4];
  const float* Wo    = (const float*)d_in[5];
  const float* Wg    = (const float*)d_in[6];
  const float* Wn    = (const float*)d_in[7];

  char* ws = (char*)d_ws;
  int*       sel  = (int*)(ws + OFF_SEL);
  float*     wgt  = (float*)(ws + OFF_WGT);
  _Float16*  KmT  = (_Float16*)(ws + OFF_KMT);
  _Float16*  Vm   = (_Float16*)(ws + OFF_VM);
  _Float16*  Qsel = (_Float16*)(ws + OFF_QSEL);
  _Float16*  attn = (_Float16*)(ws + OFF_ATTN);
  float*     osel = (float*)(ws + OFF_OSEL);
  float*     out  = (float*)d_out;

  moa_gate_kernel<<<NROW, 32, 0, stream>>>(x, noise, Wg, Wn, sel, wgt);
  moa_kv_kernel<<<dim3(NROW / 64, CC / 32, 2), 32, 0, stream>>>(x, Wk, Wv, KmT, Vm);
  moa_q_kernel<<<dim3(NROW / 64, CC / 32, EE), 32, 0, stream>>>(x, Wq, sel, Qsel);
  moa_attn_kernel<<<dim3(TT / 16, BB * HH, 2), 32, 0, stream>>>(Qsel, KmT, Vm, attn);
  moa_oproj_kernel<<<dim3(NROW / 64, CC / 32, EE * 2), 32, 0, stream>>>(attn, Wo, sel, osel);
  moa_combine_kernel<<<(NROW * CC) / 256, 256, 0, stream>>>(osel, wgt, out);
}
